// MorphologicalInteractionCore_11081015623817
// MI455X (gfx1250) — matrix operand order repaired and checked by reading, not compiled, not run on hardware
//
#include <hip/hip_runtime.h>
#include <hip/hip_bf16.h>
#include <math.h>

typedef _Float16 v16h __attribute__((ext_vector_type(16)));
typedef float    v8f  __attribute__((ext_vector_type(8)));
typedef unsigned int u32x4 __attribute__((ext_vector_type(4)));
typedef int          i32x4 __attribute__((ext_vector_type(4)));
typedef int          i32x8 __attribute__((ext_vector_type(8)));

#define NFRAMES 131072      // 128 * 1024
#define NNODES  11
#define NF      16
#define ROWSZ   176         // 11*16
#define OUT_OFF_AC 16777216 // 128*1024*128
#define AC_SIZE    23068672 // 128*1024*176
#define WS_FW16_BYTES 16777216  // rel[B,32] f32 occupies first 16 MB of ws

__device__ __forceinline__ float gelu_exact(float x) {
    return 0.5f * x * (1.0f + erff(x * 0.70710678118654752f));
}

// ---------------------------------------------------------------------------
// TDM: 1-D tensor_load_to_lds of `nelem` elements (data_size code: 0=1B,1=2B,
// 2=4B).  nelem must be < 65536 and nelem*size a multiple of 4 bytes.
// Issue from ONE wave per block; TDM ignores EXEC, one DMA per issuing wave.
// D# layout per CDNA5 ISA ch.8 (groups 2/3 zero => <=2-D tensor).
// ---------------------------------------------------------------------------
__device__ __forceinline__ void tdm_load_1d(unsigned lds_addr, const void* gptr,
                                            unsigned nelem, unsigned dsize_code) {
    const unsigned long long ga = (unsigned long long)(uintptr_t)gptr;
    u32x4 g0;
    g0[0] = 1u;                                        // count=1 (valid user D#)
    g0[1] = lds_addr;                                  // lds_addr (bytes)
    g0[2] = (unsigned)(ga & 0xffffffffu);              // global_addr[31:0]
    g0[3] = (unsigned)((ga >> 32) & 0x01ffffffu)       // global_addr[56:32]
          | 0x80000000u;                               // type=2 (bits 127:126=10b)
    i32x8 g1;
    g1[0] = (int)(dsize_code << 16);                   // wg_mask=0, data_size
    g1[1] = (int)((nelem & 0xffffu) << 16);            // tensor_dim0[15:0] @ bit48
    g1[2] = (int)(((nelem >> 16) & 0xffffu)            // tensor_dim0[31:16]
          | (1u << 16));                               // tensor_dim1 = 1
    g1[3] = (int)((nelem & 0xffffu) << 16);            // tile_dim0 @ bits 127:112
    g1[4] = 0;                                         // tile_dim1=0, tile_dim2=0
    g1[5] = (int)nelem;                                // tensor_dim0_stride[31:0]
    g1[6] = 0;                                         // stride0 hi, stride1 lo
    g1[7] = 0;                                         // stride1 hi
    const i32x4 z4 = {0, 0, 0, 0};
#if __clang_major__ >= 23
    const i32x8 z8 = {0, 0, 0, 0, 0, 0, 0, 0};
    __builtin_amdgcn_tensor_load_to_lds(g0, g1, z4, z4, z8, 0);
#else
    __builtin_amdgcn_tensor_load_to_lds(g0, g1, z4, z4, 0);
#endif
}

// ---------------------------------------------------------------------------
// Kernel 0: one-time fus_w f32 -> f16 conversion into workspace.
// ---------------------------------------------------------------------------
__global__ __launch_bounds__(256) void cvt_fw_kernel(
    const float* __restrict__ fw, _Float16* __restrict__ fw16)
{
    const int i = blockIdx.x * 256 + threadIdx.x;   // grid covers 49152 exactly
    fw16[i] = (_Float16)fw[i];
}

// ---------------------------------------------------------------------------
// Kernel 1: CanonicalGraphAdapter for agent_x and target_x.
// 16 frames per block; x-slab staged into LDS by the Tensor Data Mover;
// thread = (frame, node-row m); projection+MLP chain is thread-local.
// ---------------------------------------------------------------------------
__global__ __launch_bounds__(256) void adapter_kernel(
    const float* __restrict__ agent, const float* __restrict__ target,
    const int*   __restrict__ lab_idx,
    const float* __restrict__ proj,  const float* __restrict__ abias,
    const float* __restrict__ w1,    const float* __restrict__ b1,
    const float* __restrict__ lng,   const float* __restrict__ lnb,
    const float* __restrict__ w2,    const float* __restrict__ b2,
    float* __restrict__ ac, float* __restrict__ tc)
{
    __shared__ __align__(16) float xs[16 * ROWSZ];
    __shared__ float Ws[121], Bs[ROWSZ];
    __shared__ float W1[512], B1[32], Gs[32], Bt[32], W2[512], B2[16];

    const int bid = blockIdx.x;
    const int e   = bid >> 13;            // 0: agent, 1: target (8192 blocks each)
    const int fb  = (bid & 8191) << 4;    // frame base (16 frames, same batch b)
    const int bb  = fb >> 10;             // batch index
    const float* x = e ? target : agent;
    float*       y = e ? tc : ac;
    const int lab = lab_idx[bb];
    const int tid = threadIdx.x;

    // TDM: DMA the 16*176 f32 slab (11264 B) into LDS (wave 0 issues).
    if (tid < 32) {
        tdm_load_1d((unsigned)(uintptr_t)&xs[0], x + (size_t)fb * ROWSZ,
                    16 * ROWSZ, 2u /* 4-byte elems */);
    }
    // Small weights via regular loads (all waves).
    for (int i = tid; i < 121; i += 256) Ws[i] = proj[lab * 121 + i];
    for (int i = tid; i < ROWSZ; i += 256) Bs[i] = abias[lab * ROWSZ + i];
    for (int i = tid; i < 512; i += 256) { W1[i] = w1[i]; W2[i] = w2[i]; }
    if (tid < 32) { B1[tid] = b1[tid]; Gs[tid] = lng[tid]; Bt[tid] = lnb[tid]; }
    if (tid < 16) B2[tid] = b2[tid];
    if (tid < 32) __builtin_amdgcn_s_wait_tensorcnt(0);
    __syncthreads();

    if (tid < 16 * NNODES) {
        const int fi = tid / NNODES;
        const int m  = tid % NNODES;
        const float* xf = &xs[fi * ROWSZ];

        // projection: out[m,f] = sum_n x[n,f] * W[n,m] + bias[m,f]
        float o[NF];
        #pragma unroll
        for (int f = 0; f < NF; ++f) o[f] = Bs[m * NF + f];
        #pragma unroll
        for (int n = 0; n < NNODES; ++n) {
            const float w = Ws[n * NNODES + m];
            #pragma unroll
            for (int f = 0; f < NF; ++f) o[f] = fmaf(w, xf[n * NF + f], o[f]);
        }

        // h = out @ W1^T + b1
        float h[32];
        float mu = 0.f;
        #pragma unroll
        for (int k = 0; k < 32; ++k) {
            float acc = B1[k];
            #pragma unroll
            for (int f = 0; f < NF; ++f) acc = fmaf(o[f], W1[k * NF + f], acc);
            h[k] = acc; mu += acc;
        }
        mu *= (1.f / 32.f);
        float var = 0.f;
        #pragma unroll
        for (int k = 0; k < 32; ++k) { const float d = h[k] - mu; var = fmaf(d, d, var); }
        const float rs = rsqrtf(var * (1.f / 32.f) + 1e-5f);
        #pragma unroll
        for (int k = 0; k < 32; ++k) {
            const float z = (h[k] - mu) * rs * Gs[k] + Bt[k];
            h[k] = gelu_exact(z);
        }

        // y = h @ W2^T + b2
        float* yo = y + (size_t)(fb + fi) * ROWSZ + m * NF;
        #pragma unroll
        for (int f = 0; f < NF; ++f) {
            float acc = B2[f];
            #pragma unroll
            for (int k = 0; k < 32; ++k) acc = fmaf(h[k], W2[f * 32 + k], acc);
            yo[f] = acc;
        }
    }
}

// ---------------------------------------------------------------------------
// Kernel 2: per-frame MHA + residual layernorm + node-mean + rel MLP.
// One wave32 per frame (8 frames / 256-thread block). lane = query row i.
// Writes rel[B,32] to workspace.
// ---------------------------------------------------------------------------
__global__ __launch_bounds__(256) void mha_rel_kernel(
    const float* __restrict__ ac, const float* __restrict__ tc,
    const float* __restrict__ inW, const float* __restrict__ inB,
    const float* __restrict__ outW, const float* __restrict__ outB,
    const float* __restrict__ ng, const float* __restrict__ nb,
    const float* __restrict__ rw1, const float* __restrict__ rb1,
    const float* __restrict__ rw2, const float* __restrict__ rb2,
    float* __restrict__ rel)
{
    __shared__ float Wq[256], Wk[256], Wv[256], Wo[256];
    __shared__ float bq[16], bk[16], bv[16], bo[16], Gn[16], Bn[16];
    __shared__ float RW1[2048], RB1[64], RW2[2048], RB2[32];
    __shared__ float kbuf[8][NNODES][NF], vbuf[8][NNODES][NF];
    __shared__ float comb[8][NNODES][32];
    __shared__ float summ[8][32];
    __shared__ float r1[8][64];

    const int tid = threadIdx.x;
    for (int i = tid; i < 256; i += 256) {
        Wq[i] = inW[i]; Wk[i] = inW[256 + i]; Wv[i] = inW[512 + i]; Wo[i] = outW[i];
    }
    if (tid < 16) {
        bq[tid] = inB[tid]; bk[tid] = inB[16 + tid]; bv[tid] = inB[32 + tid];
        bo[tid] = outB[tid]; Gn[tid] = ng[tid]; Bn[tid] = nb[tid];
    }
    for (int i = tid; i < 2048; i += 256) { RW1[i] = rw1[i]; RW2[i] = rw2[i]; }
    if (tid < 64) RB1[tid] = rb1[tid];
    if (tid < 32) RB2[tid] = rb2[tid];
    __syncthreads();

    const int wv   = tid >> 5;
    const int lane = tid & 31;
    const size_t frame = (size_t)blockIdx.x * 8 + wv;

    float arow[NF], q[NF];
    if (lane < NNODES) {
        const float* ap = ac + frame * ROWSZ + lane * NF;
        const float* tp = tc + frame * ROWSZ + lane * NF;
        float trow[NF];
        #pragma unroll
        for (int f = 0; f < NF; ++f) { arow[f] = ap[f]; trow[f] = tp[f]; }
        #pragma unroll
        for (int f = 0; f < NF; ++f) {
            float qq = bq[f], kk = bk[f], vv = bv[f];
            #pragma unroll
            for (int g = 0; g < NF; ++g) {
                qq = fmaf(arow[g], Wq[f * NF + g], qq);
                kk = fmaf(trow[g], Wk[f * NF + g], kk);
                vv = fmaf(trow[g], Wv[f * NF + g], vv);
            }
            q[f] = qq;
            kbuf[wv][lane][f] = kk;
            vbuf[wv][lane][f] = vv;
        }
    }
    __syncthreads();

    if (lane < NNODES) {
        float o[NF];
        #pragma unroll
        for (int f = 0; f < NF; ++f) o[f] = 0.f;
        #pragma unroll
        for (int h = 0; h < 4; ++h) {
            float s[NNODES];
            float mx = -3.4e38f;
            #pragma unroll
            for (int j = 0; j < NNODES; ++j) {
                float d = 0.f;
                #pragma unroll
                for (int dd = 0; dd < 4; ++dd)
                    d = fmaf(q[h * 4 + dd], kbuf[wv][j][h * 4 + dd], d);
                s[j] = d * 0.5f;                    // 1/sqrt(4)
                mx = fmaxf(mx, s[j]);
            }
            float se = 0.f;
            #pragma unroll
            for (int j = 0; j < NNODES; ++j) { s[j] = expf(s[j] - mx); se += s[j]; }
            const float inv = 1.f / se;
            #pragma unroll
            for (int j = 0; j < NNODES; ++j) {
                const float w = s[j] * inv;
                #pragma unroll
                for (int dd = 0; dd < 4; ++dd)
                    o[h * 4 + dd] = fmaf(w, vbuf[wv][j][h * 4 + dd], o[h * 4 + dd]);
            }
        }
        // out projection + residual + layernorm(16)
        float r[NF];
        float mu = 0.f;
        #pragma unroll
        for (int f = 0; f < NF; ++f) {
            float a2 = bo[f];
            #pragma unroll
            for (int g = 0; g < NF; ++g) a2 = fmaf(o[g], Wo[f * NF + g], a2);
            r[f] = arow[f] + a2; mu += r[f];
        }
        mu *= (1.f / 16.f);
        float var = 0.f;
        #pragma unroll
        for (int f = 0; f < NF; ++f) { const float d = r[f] - mu; var = fmaf(d, d, var); }
        const float rs = rsqrtf(var * (1.f / 16.f) + 1e-5f);
        #pragma unroll
        for (int f = 0; f < NF; ++f) {
            comb[wv][lane][f]      = arow[f];
            comb[wv][lane][16 + f] = (r[f] - mu) * rs * Gn[f] + Bn[f];
        }
    }
    __syncthreads();

    // node-mean: lane c owns summ[c]
    {
        float s = 0.f;
        #pragma unroll
        for (int i = 0; i < NNODES; ++i) s += comb[wv][i][lane];
        summ[wv][lane] = s * (1.f / 11.f);
    }
    __syncthreads();

    // rel1 = gelu(summ @ rel_w1^T + rb1): lane computes outputs lane, lane+32
    {
        float a0 = RB1[lane], a1 = RB1[lane + 32];
        #pragma unroll
        for (int k = 0; k < 32; ++k) {
            const float sv = summ[wv][k];
            a0 = fmaf(sv, RW1[lane * 32 + k], a0);
            a1 = fmaf(sv, RW1[(lane + 32) * 32 + k], a1);
        }
        r1[wv][lane]      = gelu_exact(a0);
        r1[wv][lane + 32] = gelu_exact(a1);
    }
    __syncthreads();

    // rel = rel1 @ rel_w2^T + rb2: lane computes output lane
    {
        float a0 = RB2[lane];
        #pragma unroll
        for (int k = 0; k < 64; ++k) a0 = fmaf(r1[wv][k], RW2[lane * 64 + k], a0);
        rel[frame * 32 + lane] = a0;
    }
}

// ---------------------------------------------------------------------------
// Kernel 3: fusion GEMM with WMMA.
// out[B,128] = concat(a_c, t_c, rel)[B,384] @ fus_w^T + fus_b.
// Wave = one 16-frame M-tile; f16 inputs, f32 accumulation.
// fus_w (pre-converted f16) is staged into LDS by the Tensor Data Mover.
// ---------------------------------------------------------------------------
__device__ __forceinline__ const float* fused_src(
    const float* __restrict__ arowp, const float* __restrict__ trowp,
    const float* __restrict__ rrowp, int s)
{
    if (s < 176) return arowp + s;
    if (s < 352) return trowp + (s - 176);
    return rrowp + (s - 352);
}

__global__ __launch_bounds__(256) void fusion_gemm_kernel(
    const float* __restrict__ ac, const float* __restrict__ tc,
    const float* __restrict__ rel, const _Float16* __restrict__ fw16,
    const float* __restrict__ fb, float* __restrict__ out)
{
    // fus_w as f16, 64 output channels at a time: 64*384 halves = 48 KB LDS
    __shared__ __align__(32) _Float16 Bsh[64 * 384];

    const int tid  = threadIdx.x;
    const int wv   = tid >> 5;
    const int lane = tid & 31;
    const int mtile = blockIdx.x * 8 + wv;
    const int row16 = lane & 15;
    const int hi    = lane >> 4;
    const size_t frame = (size_t)mtile * 16 + row16;

    const float* arowp = ac  + frame * ROWSZ;
    const float* trowp = tc  + frame * ROWSZ;
    const float* rrowp = rel + frame * 32;

    // Load the whole A strip (16 frames x 384 K) for this lane, f32->f16,
    // directly into A registers, contiguous k order (B uses the same):
    //   lane<16 : K 0..15; lane>=16 : K 16..31 per chunk.
    v16h aA[12];
    #pragma unroll
    for (int c = 0; c < 12; ++c) {
        const int k0 = c * 32 + hi * 16;
        const float* p0 = fused_src(arowp, trowp, rrowp, k0);
        const float* p1 = fused_src(arowp, trowp, rrowp, k0 + 8);
        const float4 u0 = ((const float4*)p0)[0];
        const float4 u1 = ((const float4*)p0)[1];
        const float4 w0 = ((const float4*)p1)[0];
        const float4 w1 = ((const float4*)p1)[1];
        v16h a;
        a[0]  = (_Float16)u0.x; a[1]  = (_Float16)u0.y;
        a[2]  = (_Float16)u0.z; a[3]  = (_Float16)u0.w;
        a[4]  = (_Float16)u1.x; a[5]  = (_Float16)u1.y;
        a[6]  = (_Float16)u1.z; a[7]  = (_Float16)u1.w;
        a[8]  = (_Float16)w0.x; a[9]  = (_Float16)w0.y;
        a[10] = (_Float16)w0.z; a[11] = (_Float16)w0.w;
        a[12] = (_Float16)w1.x; a[13] = (_Float16)w1.y;
        a[14] = (_Float16)w1.z; a[15] = (_Float16)w1.w;
        aA[c] = a;
    }

    #pragma unroll 1
    for (int half = 0; half < 2; ++half) {
        // TDM: DMA 64 channels (24576 f16 = 48 KB) of fus_w into LDS.
        if (wv == 0) {
            tdm_load_1d((unsigned)(uintptr_t)&Bsh[0], fw16 + half * (64 * 384),
                        64 * 384, 1u /* 2-byte elems */);
            __builtin_amdgcn_s_wait_tensorcnt(0);
        }
        __syncthreads();

        v8f acc[4];
        #pragma unroll
        for (int n = 0; n < 4; ++n) {
            v8f z = {0.f, 0.f, 0.f, 0.f, 0.f, 0.f, 0.f, 0.f};
            acc[n] = z;
        }

        #pragma unroll
        for (int c = 0; c < 12; ++c) {
            const v16h a = aA[c];
            #pragma unroll
            for (int n = 0; n < 4; ++n) {
                // B layout: lane<16 -> col = lane, K 0..15 of chunk;
                //           lane>=16 -> col = lane-16, K 16..31 of chunk.
                const v16h bvec = *(const v16h*)&Bsh[(n * 16 + row16) * 384 + c * 32 + hi * 16];
                acc[n] = __builtin_amdgcn_wmma_f32_16x16x32_f16(
                    false, a, false, bvec, (short)0, acc[n], false, false);
            }
        }

        // D layout: col = lane&15 within N-tile; rows = vgpr (+8 for lanes>=16)
        const int rbase = mtile * 16 + hi * 8;
        #pragma unroll
        for (int n = 0; n < 4; ++n) {
            const int col  = half * 64 + n * 16 + row16;
            const float bias = fb[col];
            #pragma unroll
            for (int i = 0; i < 8; ++i)
                out[(size_t)(rbase + i) * 128 + col] = acc[n][i] + bias;
        }
        __syncthreads();   // protect Bsh before next half restages it
    }
}

// ---------------------------------------------------------------------------
extern "C" void kernel_launch(void* const* d_in, const int* in_sizes, int n_in,
                              void* d_out, int out_size, void* d_ws, size_t ws_size,
                              hipStream_t stream) {
    const float* agent     = (const float*)d_in[0];
    const float* target    = (const float*)d_in[1];
    const int*   lab_idx   = (const int*)  d_in[2];
    const float* projection= (const float*)d_in[3];
    const float* abias     = (const float*)d_in[4];
    const float* ref_w1    = (const float*)d_in[5];
    const float* ref_b1    = (const float*)d_in[6];
    const float* ref_ln_g  = (const float*)d_in[7];
    const float* ref_ln_b  = (const float*)d_in[8];
    const float* ref_w2    = (const float*)d_in[9];
    const float* ref_b2    = (const float*)d_in[10];
    const float* in_proj_w = (const float*)d_in[11];
    const float* in_proj_b = (const float*)d_in[12];
    const float* out_proj_w= (const float*)d_in[13];
    const float* out_proj_b= (const float*)d_in[14];
    const float* norm_g    = (const float*)d_in[15];
    const float* norm_b    = (const float*)d_in[16];
    const float* rel_w1    = (const float*)d_in[17];
    const float* rel_b1    = (const float*)d_in[18];
    const float* rel_w2    = (const float*)d_in[19];
    const float* rel_b2    = (const float*)d_in[20];
    const float* fus_w     = (const float*)d_in[21];
    const float* fus_b     = (const float*)d_in[22];

    float*     out  = (float*)d_out;                         // [B,128]
    float*     ac   = out + OUT_OFF_AC;                      // [B,176]
    float*     tc   = ac + AC_SIZE;                          // [B,176]
    float*     rel  = (float*)d_ws;                          // [B,32] (16 MB)
    _Float16*  fw16 = (_Float16*)((char*)d_ws + WS_FW16_BYTES); // 96 KB

    cvt_fw_kernel<<<192, 256, 0, stream>>>(fus_w, fw16);     // 192*256 = 49152

    adapter_kernel<<<16384, 256, 0, stream>>>(
        agent, target, lab_idx, projection, abias,
        ref_w1, ref_b1, ref_ln_g, ref_ln_b, ref_w2, ref_b2, ac, tc);

    mha_rel_kernel<<<16384, 256, 0, stream>>>(
        ac, tc, in_proj_w, in_proj_b, out_proj_w, out_proj_b,
        norm_g, norm_b, rel_w1, rel_b1, rel_w2, rel_b2, rel);

    fusion_gemm_kernel<<<1024, 256, 0, stream>>>(ac, tc, rel, fw16, fus_b, out);
}
